// _NonLocalBlockND_7327214207169
// MI455X (gfx1250) — matrix operand order repaired and checked by reading, not compiled, not run on hardware
//
#include <hip/hip_runtime.h>
#include <hip/hip_bf16.h>

// ---------------------------------------------------------------------------
// NonLocal block (self-attention) for MI455X / gfx1250.
// All GEMM stages use V_WMMA_F32_16X16X32_BF16 (bf16 in, f32 accumulate).
// Attention is fully fused (flash-style): scores -> online softmax -> context
// never touch HBM; per-wave P tiles are transposed D-layout->A-layout via LDS.
// Fragment layouts as used here (A and B share one contiguous k order):
//   A 16x32 bf16 : lane row = lane&15 ; K = koff + 0..15,
//                  koff = (lane>>4)*16 -> two contiguous 16B loads
//   B 32x16 bf16 : lane col = lane&15 ; K = (lane>>4)*16 + 0..15 -> one 32B load
//   D 16x16 f32  : lane col = lane&15 ; rows = (lane>>4)*8 + r, r = VGPR 0..7
// ---------------------------------------------------------------------------

typedef __attribute__((ext_vector_type(16))) __bf16 v16bf;
typedef __attribute__((ext_vector_type(8)))  __bf16 v8bf;
typedef __attribute__((ext_vector_type(8)))  float  v8f;

constexpr int Bb = 4;                      // batch
constexpr int Cc = 256;                    // channels
constexpr int Ci = 128;                    // inner channels
constexpr int Tt = 16, Hh = 28, Ww = 28;
constexpr int Nn = Tt * Hh * Ww;           // 12544 queries
constexpr int Mm = (Tt/2)*(Hh/2)*(Ww/2);   // 1568 pooled keys
constexpr int NT = Nn / 16;                // 784 n-tiles
constexpr float BN_EPS = 1e-5f;

static __device__ __forceinline__ __bf16 f2bf(float f) {
  union { float f; unsigned u; } v; v.f = f;
  unsigned r = v.u + 0x7FFFu + ((v.u >> 16) & 1u);     // round-to-nearest-even
  union { unsigned short s; __bf16 b; } o; o.s = (unsigned short)(r >> 16);
  return o.b;
}
static __device__ __forceinline__ float bf2f(__bf16 b) {
  union { __bf16 b; unsigned short s; } i; i.b = b;
  union { unsigned u; float f; } o; o.u = ((unsigned)i.s) << 16;
  return o.f;
}

// A-fragment: rowk points at (row base + current kbase); lane supplies koff.
static __device__ __forceinline__ v16bf load_a_frag(const __bf16* rowk, int lane) {
  int koff = (lane >> 4) * 16;
  v8bf lo = *(const v8bf*)(rowk + koff);        // K = koff .. koff+7
  v8bf hi = *(const v8bf*)(rowk + koff + 8);    // K = koff+8 .. koff+15
  return __builtin_shufflevector(lo, hi, 0,1,2,3,4,5,6,7,8,9,10,11,12,13,14,15);
}

// ---------------------------------------------------------------- elementwise
__global__ void k_f2bf(const float* __restrict__ src, __bf16* __restrict__ dst, int n) {
  int i = blockIdx.x * 256 + threadIdx.x;
  if (i < n) dst[i] = f2bf(src[i]);
}

// x (b,c,n) f32  ->  xbT (b,n,c) bf16   [B-operand layout for the QKV convs]
__global__ void k_pack_x(const float* __restrict__ x, __bf16* __restrict__ xbT) {
  int i = blockIdx.x * 256 + threadIdx.x;            // exactly B*C*N threads
  int n = i % Nn; int c = (i / Nn) % Cc; int b = i / (Nn * Cc);
  xbT[((size_t)b * Nn + n) * Cc + c] = f2bf(x[i]);
}

// -------------------------------------------------------------- QKV 1x1 conv
__global__ void k_conv_qkv(const __bf16* __restrict__ xbT,
                           const __bf16* __restrict__ wg,
                           const __bf16* __restrict__ wth,
                           const __bf16* __restrict__ wph,
                           const float* __restrict__ bg,
                           const float* __restrict__ bth,
                           const float* __restrict__ bph,
                           __bf16* __restrict__ thetaT,    // (b,n,ci)
                           __bf16* __restrict__ g_full,    // (b,ci,n)
                           __bf16* __restrict__ phi_full)  // (b,ci,n)
{
  int lane = threadIdx.x & 31;
  int tile = blockIdx.x * 8 + (threadIdx.x >> 5);
  int mat   = tile % 3;  tile /= 3;
  int rtile = tile % 8;  tile /= 8;          // Ci/16 row tiles
  int ntile = tile % NT; int b = tile / NT;
  const __bf16* wmat = (mat == 0) ? wg : (mat == 1) ? wth : wph;
  const float*  bias = (mat == 0) ? bg : (mat == 1) ? bth : bph;
  int o0 = rtile * 16, n0 = ntile * 16;

  const __bf16* arow = wmat + (size_t)(o0 + (lane & 15)) * Cc;
  const __bf16* bcol = xbT + ((size_t)b * Nn + n0 + (lane & 15)) * Cc + ((lane >> 4) * 16);
  v8f acc = {};
#pragma unroll
  for (int kb = 0; kb < Cc; kb += 32) {
    v16bf a  = load_a_frag(arow + kb, lane);
    v16bf bb = *(const v16bf*)(bcol + kb);
    acc = __builtin_amdgcn_wmma_f32_16x16x32_bf16(false, a, false, bb, (short)0, acc, false, false);
  }
  int col_n = n0 + (lane & 15);
  int rbase = o0 + ((lane >> 4) * 8);
#pragma unroll
  for (int r = 0; r < 8; ++r) {
    int o = rbase + r;
    float v = acc[r] + bias[o];
    if (mat == 1)      thetaT [((size_t)b * Nn + col_n) * Ci + o] = f2bf(v);
    else if (mat == 0) g_full [((size_t)b * Ci + o) * Nn + col_n] = f2bf(v);
    else               phi_full[((size_t)b * Ci + o) * Nn + col_n] = f2bf(v);
  }
}

// ------------------------------------------------------------- MaxPool3d(2)
__global__ void k_pool(const __bf16* __restrict__ g_full,
                       const __bf16* __restrict__ phi_full,
                       __bf16* __restrict__ g_p,     // (b,ci,m)
                       __bf16* __restrict__ phiT)    // (b,m,ci)
{
  int i = blockIdx.x * 256 + threadIdx.x;            // exactly B*Ci*M threads
  int m = i % Mm; int ci = (i / Mm) % Ci; int b = i / (Mm * Ci);
  int tp = m / 196, rr = m % 196, hp = rr / 14, wp = rr % 14;
  size_t base = ((size_t)b * Ci + ci) * Nn;
  float mg = -3.0e38f, mp = -3.0e38f;
#pragma unroll
  for (int dt = 0; dt < 2; ++dt)
#pragma unroll
    for (int dh = 0; dh < 2; ++dh)
#pragma unroll
      for (int dw = 0; dw < 2; ++dw) {
        int n = (2*tp+dt)*784 + (2*hp+dh)*28 + (2*wp+dw);
        mg = fmaxf(mg, bf2f(g_full[base + n]));
        mp = fmaxf(mp, bf2f(phi_full[base + n]));
      }
  g_p [((size_t)b * Ci + ci) * Mm + m]  = f2bf(mg);
  phiT[((size_t)b * Mm + m) * Ci + ci]  = f2bf(mp);
}

// ---------------------------------------------------- fused flash attention
// One wave owns 16 query rows and the full 16x128 context accumulator.
// Streams M in chunks of 32 keys: 8 score WMMAs (K=128), online softmax with
// cross-lane row reductions, P transposed via LDS, 8 context WMMAs (K=32).
// phiT/g_p are 400KB/batch -> resident in the 192MB L2 across all 784 tiles.
__global__ void __launch_bounds__(256) k_flash_attn(
    const __bf16* __restrict__ thetaT,  // (b,n,ci)
    const __bf16* __restrict__ phiT,    // (b,m,ci)
    const __bf16* __restrict__ g_p,     // (b,ci,m)
    __bf16* __restrict__ yT)            // (b,n,ci)
{
  __shared__ __bf16 pbuf[8][16][32];                  // 8KB: P tile per wave
  int lane = threadIdx.x & 31;
  int w    = threadIdx.x >> 5;
  int tile = blockIdx.x * 8 + w;
  int nt = tile % NT; int b = tile / NT;
  int n0 = nt * 16;
  int lr = lane & 15;                                 // row (A/D) or col (B/D)
  int lh = lane >> 4;                                 // lane half
  int rbase = lh * 8;                                 // D-layout row base

  // theta A-fragments for this query tile, kept in registers (K = Ci = 128)
  const __bf16* arow = thetaT + ((size_t)b * Nn + n0 + lr) * Ci;
  v16bf aTh[4];
#pragma unroll
  for (int kb = 0; kb < 4; ++kb) aTh[kb] = load_a_frag(arow + kb * 32, lane);

  float runmax[8], runsum[8];
  v8f Y[8];
  v8f vzero = {};
#pragma unroll
  for (int r = 0; r < 8; ++r) { runmax[r] = -3.0e38f; runsum[r] = 0.f; }
#pragma unroll
  for (int t = 0; t < 8; ++t) Y[t] = vzero;

  for (int m0 = 0; m0 < Mm; m0 += 32) {               // 49 key chunks
    // ---- scores S0 (keys m0..m0+15), S1 (keys m0+16..m0+31)
    const __bf16* bcol0 = phiT + ((size_t)b * Mm + m0 + lr) * Ci + lh * 16;
    const __bf16* bcol1 = bcol0 + (size_t)16 * Ci;
    v8f S0 = vzero, S1 = vzero;
#pragma unroll
    for (int kb = 0; kb < 4; ++kb) {
      v16bf b0 = *(const v16bf*)(bcol0 + kb * 32);
      v16bf b1 = *(const v16bf*)(bcol1 + kb * 32);
      S0 = __builtin_amdgcn_wmma_f32_16x16x32_bf16(false, aTh[kb], false, b0, (short)0, S0, false, false);
      S1 = __builtin_amdgcn_wmma_f32_16x16x32_bf16(false, aTh[kb], false, b1, (short)0, S1, false, false);
    }
    // ---- online softmax: row reductions across the 16 column lanes
    float p0[8], p1[8], sc[8];
#pragma unroll
    for (int r = 0; r < 8; ++r) {
      float cm = fmaxf(S0[r], S1[r]);
      cm = fmaxf(cm, __shfl_xor(cm, 1));
      cm = fmaxf(cm, __shfl_xor(cm, 2));
      cm = fmaxf(cm, __shfl_xor(cm, 4));
      cm = fmaxf(cm, __shfl_xor(cm, 8));
      float nm = fmaxf(runmax[r], cm);
      sc[r] = __expf(runmax[r] - nm);
      runmax[r] = nm;
      p0[r] = __expf(S0[r] - nm);
      p1[r] = __expf(S1[r] - nm);
      float rs = p0[r] + p1[r];
      rs += __shfl_xor(rs, 1);
      rs += __shfl_xor(rs, 2);
      rs += __shfl_xor(rs, 4);
      rs += __shfl_xor(rs, 8);
      runsum[r] = runsum[r] * sc[r] + rs;
    }
    // ---- rescale running context
#pragma unroll
    for (int t = 0; t < 8; ++t)
#pragma unroll
      for (int r = 0; r < 8; ++r) Y[t][r] *= sc[r];
    // ---- P: D-layout -> A-layout transpose through LDS (wave-private tile)
#pragma unroll
    for (int r = 0; r < 8; ++r) {
      pbuf[w][rbase + r][lr]      = f2bf(p0[r]);
      pbuf[w][rbase + r][lr + 16] = f2bf(p1[r]);
    }
    asm volatile("s_wait_dscnt 0x0" ::: "memory");    // DS in-order fence (intra-wave)
    int koff = lh * 16;
    v8bf lo = *(const v8bf*)&pbuf[w][lr][koff];
    v8bf hi = *(const v8bf*)&pbuf[w][lr][koff + 8];
    v16bf aP = __builtin_shufflevector(lo, hi, 0,1,2,3,4,5,6,7,8,9,10,11,12,13,14,15);
    // ---- context accumulate: Y[t] += P(16x32) @ G(32x16) per ci-tile
#pragma unroll
    for (int t = 0; t < 8; ++t) {
      const __bf16* gcol = g_p + ((size_t)b * Ci + t * 16 + lr) * Mm + m0 + lh * 16;
      v16bf bG = *(const v16bf*)gcol;
      Y[t] = __builtin_amdgcn_wmma_f32_16x16x32_bf16(false, aP, false, bG, (short)0, Y[t], false, false);
    }
  }
  // ---- normalize and store yT (b,n,ci)
#pragma unroll
  for (int t = 0; t < 8; ++t) {
#pragma unroll
    for (int r = 0; r < 8; ++r) {
      yT[((size_t)b * Nn + n0 + rbase + r) * Ci + t * 16 + lr] =
          f2bf(Y[t][r] / runsum[r]);
    }
  }
}

// ------------------------------- output conv + bias + BatchNorm + residual
__global__ void k_out_bn(const __bf16* __restrict__ wW,
                         const __bf16* __restrict__ yT,
                         const float* __restrict__ bW,
                         const float* __restrict__ gamma,
                         const float* __restrict__ beta,
                         const float* __restrict__ mean,
                         const float* __restrict__ var,
                         const float* __restrict__ x,
                         float* __restrict__ out)
{
  int lane = threadIdx.x & 31;
  int tile = blockIdx.x * 8 + (threadIdx.x >> 5);
  int ct = tile % 16; tile /= 16;           // C/16 row tiles
  int nt = tile % NT; int b = tile / NT;
  int co0 = ct * 16, n0 = nt * 16;
  const __bf16* arow = wW + (size_t)(co0 + (lane & 15)) * Ci;
  const __bf16* bcol = yT + ((size_t)b * Nn + n0 + (lane & 15)) * Ci + ((lane >> 4) * 16);
  v8f acc = {};
#pragma unroll
  for (int kb = 0; kb < Ci; kb += 32) {
    v16bf a  = load_a_frag(arow + kb, lane);
    v16bf bb = *(const v16bf*)(bcol + kb);
    acc = __builtin_amdgcn_wmma_f32_16x16x32_bf16(false, a, false, bb, (short)0, acc, false, false);
  }
  int n = n0 + (lane & 15);
  int rbase = co0 + ((lane >> 4) * 8);
#pragma unroll
  for (int r = 0; r < 8; ++r) {
    int co = rbase + r;
    float scale = gamma[co] * rsqrtf(var[co] + BN_EPS);
    size_t idx = ((size_t)b * Cc + co) * Nn + n;
    out[idx] = (acc[r] + bW[co] - mean[co]) * scale + beta[co] + x[idx];
  }
}

// ---------------------------------------------------------------------------
extern "C" void kernel_launch(void* const* d_in, const int* in_sizes, int n_in,
                              void* d_out, int out_size, void* d_ws, size_t ws_size,
                              hipStream_t stream) {
  (void)in_sizes; (void)n_in; (void)out_size; (void)ws_size;
  const float* x       = (const float*)d_in[0];
  const float* w_g     = (const float*)d_in[1];
  const float* b_g     = (const float*)d_in[2];
  const float* w_theta = (const float*)d_in[3];
  const float* b_theta = (const float*)d_in[4];
  const float* w_phi   = (const float*)d_in[5];
  const float* b_phi   = (const float*)d_in[6];
  const float* w_W     = (const float*)d_in[7];
  const float* b_W     = (const float*)d_in[8];
  const float* bn_g    = (const float*)d_in[9];
  const float* bn_b    = (const float*)d_in[10];
  const float* bn_m    = (const float*)d_in[11];
  const float* bn_v    = (const float*)d_in[12];
  float* out = (float*)d_out;

  // workspace carve-up (~80 MB total after flash fusion)
  char* p = (char*)d_ws;
  auto take = [&](size_t bytes) { char* r = p; p += (bytes + 255) & ~(size_t)255; return r; };
  __bf16* xbT      = (__bf16*)take((size_t)Bb * Nn * Cc * 2);   // 25.7 MB
  __bf16* wg_bf    = (__bf16*)take((size_t)Ci * Cc * 2);
  __bf16* wth_bf   = (__bf16*)take((size_t)Ci * Cc * 2);
  __bf16* wph_bf   = (__bf16*)take((size_t)Ci * Cc * 2);
  __bf16* wW_bf    = (__bf16*)take((size_t)Cc * Ci * 2);
  __bf16* thetaT   = (__bf16*)take((size_t)Bb * Nn * Ci * 2);   // 12.8 MB
  __bf16* g_full   = (__bf16*)take((size_t)Bb * Ci * Nn * 2);   // 12.8 MB
  __bf16* phi_full = (__bf16*)take((size_t)Bb * Ci * Nn * 2);   // 12.8 MB
  __bf16* g_p      = (__bf16*)take((size_t)Bb * Ci * Mm * 2);   // 1.6 MB
  __bf16* phiT     = (__bf16*)take((size_t)Bb * Mm * Ci * 2);   // 1.6 MB
  __bf16* yT       = (__bf16*)take((size_t)Bb * Nn * Ci * 2);   // 12.8 MB

  // 1. pack weights and input to bf16
  k_f2bf<<<(Ci * Cc + 255) / 256, 256, 0, stream>>>(w_g,     wg_bf,  Ci * Cc);
  k_f2bf<<<(Ci * Cc + 255) / 256, 256, 0, stream>>>(w_theta, wth_bf, Ci * Cc);
  k_f2bf<<<(Ci * Cc + 255) / 256, 256, 0, stream>>>(w_phi,   wph_bf, Ci * Cc);
  k_f2bf<<<(Cc * Ci + 255) / 256, 256, 0, stream>>>(w_W,     wW_bf,  Cc * Ci);
  k_pack_x<<<(Bb * Cc * Nn) / 256, 256, 0, stream>>>(x, xbT);

  // 2. QKV 1x1 convs (WMMA): B*784*8*3 = 75264 tiles, 8 waves/block
  k_conv_qkv<<<(Bb * NT * 8 * 3) / 8, 256, 0, stream>>>(
      xbT, wg_bf, wth_bf, wph_bf, b_g, b_theta, b_phi, thetaT, g_full, phi_full);

  // 3. MaxPool3d(2) on g and phi
  k_pool<<<(Bb * Ci * Mm) / 256, 256, 0, stream>>>(g_full, phi_full, g_p, phiT);

  // 4. fused attention (scores + softmax + context), one wave per 16 queries
  k_flash_attn<<<(Bb * NT) / 8, 256, 0, stream>>>(thetaT, phiT, g_p, yT);

  // 5. output conv + BN + residual (WMMA): B*784*16 tiles
  k_out_bn<<<(Bb * NT * 16) / 8, 256, 0, stream>>>(
      wW_bf, yT, b_W, bn_g, bn_b, bn_m, bn_v, x, out);
}
